// SelfAttention_67113158967636
// MI455X (gfx1250) — matrix operand order repaired and checked by reading, not compiled, not run on hardware
//
#include <hip/hip_runtime.h>
#include <hip/hip_bf16.h>
#include <math.h>

// MI455X / gfx1250, wave32. Core compute: v_wmma_f32_16x16x32_f16 (f16 in, f32 acc).
//
// Roofline: ~41 GFLOP total, all operands (Q/K/V f16 ~10 MB) L2-resident ->
// compute-bound on WMMA. All hot-loop global accesses are contiguous 16/32-byte
// vector loads; fp32->f16 conversion is hoisted into one-time prep kernels.
//
// Workspace layout:
//   Qb  : B*N*32  f16   Kb : B*N*32 f16   Vt : B*C*N f16 (V transposed [b][c][n])
//   Mr  : B*N f32 (row max)   IS : B*N f32 (1/rowsum)
//   xh  : B*N*C f16 (x pre-converted)
//   Wqt/Wkt : 32*256 f16 (W^T)   Wvt : 256*256 f16 (W^T)
// total ~18.8 MB

typedef __attribute__((ext_vector_type(16))) _Float16 v16h;
typedef __attribute__((ext_vector_type(8)))  _Float16 v8h;
typedef __attribute__((ext_vector_type(4)))  _Float16 v4h;
typedef __attribute__((ext_vector_type(8)))  float    v8f;

#define WMMA16(a, b, c) \
  __builtin_amdgcn_wmma_f32_16x16x32_f16(false, (a), false, (b), (short)0, (c), false, false)

// ---- WMMA operand loaders (contiguous k = hi*16 + t, same order in A and B) ----

// A (16x32 f16): lane (lo,hi): row = row0+lo; halves 0..15 = K hi*16+0..15.
// rowptr points at [row][k0].
static __device__ __forceinline__ v16h load_a16x32(const _Float16* __restrict__ rowptr, int hi) {
  const v8h p0 = *(const v8h*)(rowptr + hi * 16);
  const v8h p1 = *(const v8h*)(rowptr + hi * 16 + 8);
  v16h a;
#pragma unroll
  for (int t = 0; t < 8; ++t) { a[t] = p0[t]; a[8 + t] = p1[t]; }
  return a;
}

// B (32x16 f16) where B[k][n] = src[n][k0+k] (src row-major, ld = ldm):
// lane (lo,hi): col, halves t = K hi*16+t -> one contiguous 32B load.
static __device__ __forceinline__ v16h load_bt32(const _Float16* __restrict__ src,
                                                 int col, int ldm, int k0, int hi) {
  return *(const v16h*)(src + (size_t)col * ldm + k0 + hi * 16);
}

// =====================================================================
// Prep 0a: weights -> f16 transposed (Wt[n][k] = W[k][n]); tiny, one-shot.
// =====================================================================
__global__ void prep_weights(const float* __restrict__ Wq, const float* __restrict__ Wk,
                             const float* __restrict__ Wv,
                             _Float16* __restrict__ Wqt, _Float16* __restrict__ Wkt,
                             _Float16* __restrict__ Wvt) {
  const int C = 256, D = 32;
  const int idx = blockIdx.x * blockDim.x + threadIdx.x;
  if (idx < C * D) {
    const int k = idx / D, c = idx % D;
    Wqt[(size_t)c * C + k] = (_Float16)Wq[idx];
    Wkt[(size_t)c * C + k] = (_Float16)Wk[idx];
  }
  if (idx < C * C) {
    const int k = idx / C, c = idx % C;
    Wvt[(size_t)c * C + k] = (_Float16)Wv[idx];
  }
}

// =====================================================================
// Prep 0b: x -> f16 (straight copy-convert), 4 elems/thread, b128 in / b64 out.
// =====================================================================
__global__ void prep_x(const float* __restrict__ x, _Float16* __restrict__ xh) {
  const size_t i = ((size_t)blockIdx.x * blockDim.x + threadIdx.x) * 4;
  const float4 v = *(const float4*)(x + i);
  v4h o;
  o[0] = (_Float16)v.x; o[1] = (_Float16)v.y; o[2] = (_Float16)v.z; o[3] = (_Float16)v.w;
  *(v4h*)(xh + i) = o;
}

// =====================================================================
// Kernel 1: QKV projection.  One wave per 16x16 output tile, K-loop over C=256.
// ot 0..1 -> Q, 2..3 -> K, 4..19 -> V (stored transposed).
// All operand loads are contiguous 16/32-byte vector loads.
// =====================================================================
__global__ void qkv_kernel(const _Float16* __restrict__ xh,
                           const _Float16* __restrict__ Wqt, const float* __restrict__ bq,
                           const _Float16* __restrict__ Wkt, const float* __restrict__ bk,
                           const _Float16* __restrict__ Wvt, const float* __restrict__ bv,
                           _Float16* __restrict__ Qb, _Float16* __restrict__ Kb,
                           _Float16* __restrict__ Vt) {
  const int N = 4096, C = 256, D = 32;
  const int wave = blockIdx.x * 2 + (threadIdx.x >> 5);
  const int lane = threadIdx.x & 31, lo = lane & 15, hi = lane >> 4;
  const int ot = wave % 20;
  const int nt = (wave / 20) % (N / 16);
  const int b  = wave / (20 * (N / 16));
  const int n0 = nt * 16;

  const _Float16* Wt;  const float* bias;  int col0;
  if (ot < 2)      { Wt = Wqt; bias = bq; col0 = ot * 16; }
  else if (ot < 4) { Wt = Wkt; bias = bk; col0 = (ot - 2) * 16; }
  else             { Wt = Wvt; bias = bv; col0 = (ot - 4) * 16; }

  const _Float16* xrow = xh + ((size_t)b * N + n0 + lo) * C;  // A row for this lane
  const _Float16* wrow = Wt + (size_t)(col0 + lo) * C;        // B column for this lane

  v8f acc = {};
#pragma unroll
  for (int kk = 0; kk < C / 32; ++kk) {
    const int k0 = kk * 32;
    const v16h a  = load_a16x32(xrow + k0, hi);
    const v16h bm = *(const v16h*)(wrow + k0 + hi * 16);
    acc = WMMA16(a, bm, acc);
  }

  const float bcol = bias[col0 + lo];
  if (ot < 4) {
    _Float16* dst = (ot < 2 ? Qb : Kb) + (size_t)b * N * D;
#pragma unroll
    for (int r = 0; r < 8; ++r)
      dst[(size_t)(n0 + hi * 8 + r) * D + col0 + lo] = (_Float16)(acc[r] + bcol);
  } else {
    // transposed store: Vt[b][c][n]; 8 consecutive n per lane -> one 16B store
    _Float16* dst = Vt + ((size_t)b * C + col0 + lo) * N + n0 + hi * 8;
    v8h o;
#pragma unroll
    for (int r = 0; r < 8; ++r) o[r] = (_Float16)(acc[r] + bcol);
    *(v8h*)dst = o;
  }
}

// =====================================================================
// Kernel 2: per-row softmax statistics of energy = Q K^T.
// One wave per 16-row i-tile; d=32 -> one WMMA per 16x16 energy tile.
// =====================================================================
__global__ void row_stats_kernel(const _Float16* __restrict__ Qb,
                                 const _Float16* __restrict__ Kb,
                                 float* __restrict__ Mr, float* __restrict__ IS) {
  const int N = 4096, D = 32;
  const int wave = blockIdx.x * 2 + (threadIdx.x >> 5);
  const int lane = threadIdx.x & 31, lo = lane & 15, hi = lane >> 4;
  const int it = wave % (N / 16);
  const int b  = wave / (N / 16);
  const int i0 = it * 16;

  const _Float16* Qbase = Qb + (size_t)b * N * D;
  const _Float16* Kbase = Kb + (size_t)b * N * D;

  const v16h a = load_a16x32(Qbase + (size_t)(i0 + lo) * D, hi);  // fixed across j

  float m[8], s[8];
#pragma unroll
  for (int r = 0; r < 8; ++r) { m[r] = -INFINITY; s[r] = 0.0f; }

#pragma unroll 2
  for (int j0 = 0; j0 < N; j0 += 16) {
    const v16h bm = load_bt32(Kbase, j0 + lo, D, 0, hi);
    v8f z = {};
    const v8f e = WMMA16(a, bm, z);   // lane holds rows i0+hi*8+r, col j0+lo
#pragma unroll
    for (int r = 0; r < 8; ++r) {
      const float nm = fmaxf(m[r], e[r]);
      s[r] = s[r] * __expf(m[r] - nm) + __expf(e[r] - nm);
      m[r] = nm;
    }
  }
  // merge the 16 lane-partials (j strided mod 16) within each hi half
#pragma unroll
  for (int mask = 1; mask < 16; mask <<= 1) {
#pragma unroll
    for (int r = 0; r < 8; ++r) {
      const float om = __shfl_xor(m[r], mask, 32);
      const float os = __shfl_xor(s[r], mask, 32);
      const float nm = fmaxf(m[r], om);
      s[r] = s[r] * __expf(m[r] - nm) + os * __expf(om - nm);
      m[r] = nm;
    }
  }
  if (lo == 0) {
#pragma unroll
    for (int r = 0; r < 8; ++r) {
      const int row = i0 + hi * 8 + r;
      Mr[(size_t)b * N + row] = m[r];
      IS[(size_t)b * N + row] = 1.0f / s[r];
    }
  }
}

// =====================================================================
// Kernel 3: out^T[c][j] = sum_i V^T[c][i] * P[i][j],
//   P[i][j] = exp(q_i.k_j - m_i) / s_i, recomputed on the fly via WMMA.
// One wave per (j-tile, 128-channel group); 8 accumulator tiles per wave.
// Final store fuses: out = gamma * attn_out + x.
// =====================================================================
__global__ void attn_out_kernel(const _Float16* __restrict__ Qb,
                                const _Float16* __restrict__ Kb,
                                const _Float16* __restrict__ Vt,
                                const float* __restrict__ Mr,
                                const float* __restrict__ IS,
                                const float* __restrict__ x,
                                const float* __restrict__ gamma_p,
                                float* __restrict__ out) {
  const int N = 4096, C = 256, D = 32, CT = 8;
  const int wave = blockIdx.x * 2 + (threadIdx.x >> 5);
  const int lane = threadIdx.x & 31, lo = lane & 15, hi = lane >> 4;
  const int cg = wave % 2;
  const int jt = (wave / 2) % (N / 16);
  const int b  = wave / (2 * (N / 16));
  const int j0 = jt * 16;
  const int cbase = cg * 128;

  const _Float16* Qbase = Qb + (size_t)b * N * D;
  const _Float16* Kbase = Kb + (size_t)b * N * D;
  const _Float16* Vbase = Vt + (size_t)b * C * N;
  const float* Mb = Mr + (size_t)b * N;
  const float* Sb = IS + (size_t)b * N;

  // Energy B-operand depends only on the j-tile: hoisted out of the i-loop.
  const v16h bK = load_bt32(Kbase, j0 + lo, D, 0, hi);

  v8f acc[CT];
#pragma unroll
  for (int ct = 0; ct < CT; ++ct) acc[ct] = (v8f){};

  for (int i0 = 0; i0 < N; i0 += 32) {
    if (i0 + 32 < N)  // gfx1250: global_prefetch_b8 for next Q rows
      __builtin_prefetch(Qbase + (size_t)(i0 + 32 + lo) * D, 0, 1);

    const v16h aq0 = load_a16x32(Qbase + (size_t)(i0 + lo) * D, hi);
    const v16h aq1 = load_a16x32(Qbase + (size_t)(i0 + 16 + lo) * D, hi);
    v8f z = {};
    const v8f e0 = WMMA16(aq0, bK, z);  // rows i0+hi*8+r
    const v8f e1 = WMMA16(aq1, bK, z);  // rows i0+16+hi*8+r

    // per-row stats: 8 consecutive floats -> aligned 32B vector loads
    const v8f m0v = *(const v8f*)(Mb + i0 + hi * 8);
    const v8f m1v = *(const v8f*)(Mb + i0 + 16 + hi * 8);
    const v8f s0v = *(const v8f*)(Sb + i0 + hi * 8);
    const v8f s1v = *(const v8f*)(Sb + i0 + 16 + hi * 8);

    float p0[8], p1[8];
#pragma unroll
    for (int r = 0; r < 8; ++r) {
      p0[r] = __expf(e0[r] - m0v[r]) * s0v[r];
      p1[r] = __expf(e1[r] - m1v[r]) * s1v[r];
    }

    // Assemble P as a B-operand (lane needs rows i0 + hi*16 + 0..15, col j0+lo).
    // Exchange the "other half" rows with the partner lane (lane ^ 16).
    v16h bp;
#pragma unroll
    for (int r = 0; r < 8; ++r) {
      const float send = hi ? p0[r] : p1[r];
      const float recv = __shfl_xor(send, 16, 32);
      const float own  = hi ? p1[r] : p0[r];
      bp[r]     = (_Float16)(hi ? recv : own);  // K = hi*16 + r
      bp[8 + r] = (_Float16)(hi ? own  : recv); // K = hi*16 + 8 + r
    }

#pragma unroll
    for (int ct = 0; ct < CT; ++ct) {
      const _Float16* vp = Vbase + (size_t)(cbase + ct * 16 + lo) * N + i0;
      const v16h av = load_a16x32(vp, hi);        // A = V^T tile (c x i)
      acc[ct] = WMMA16(av, bp, acc[ct]);          // out^T += V^T * P
    }
  }

  const float g = gamma_p[0];
#pragma unroll
  for (int ct = 0; ct < CT; ++ct) {
#pragma unroll
    for (int r = 0; r < 8; ++r) {
      const int c = cbase + ct * 16 + hi * 8 + r;
      const size_t idx = ((size_t)b * N + j0 + lo) * C + c;
      out[idx] = g * acc[ct][r] + x[idx];
    }
  }
}

// =====================================================================
extern "C" void kernel_launch(void* const* d_in, const int* in_sizes, int n_in,
                              void* d_out, int out_size, void* d_ws, size_t ws_size,
                              hipStream_t stream) {
  const int B = 4, N = 4096, C = 256, D = 32;

  const float* x  = (const float*)d_in[0];
  const float* Wq = (const float*)d_in[1];
  const float* bq = (const float*)d_in[2];
  const float* Wk = (const float*)d_in[3];
  const float* bk = (const float*)d_in[4];
  const float* Wv = (const float*)d_in[5];
  const float* bv = (const float*)d_in[6];
  const float* gm = (const float*)d_in[7];
  float* out = (float*)d_out;

  // carve workspace (all offsets keep >=16B alignment)
  _Float16* Qb  = (_Float16*)d_ws;                  // B*N*D
  _Float16* Kb  = Qb + (size_t)B * N * D;           // B*N*D
  _Float16* Vt  = Kb + (size_t)B * N * D;           // B*C*N
  float*    Mr  = (float*)(Vt + (size_t)B * C * N); // B*N
  float*    IS  = Mr + (size_t)B * N;               // B*N
  _Float16* xh  = (_Float16*)(IS + (size_t)B * N);  // B*N*C
  _Float16* Wqt = xh + (size_t)B * N * C;           // D*C
  _Float16* Wkt = Wqt + (size_t)D * C;              // D*C
  _Float16* Wvt = Wkt + (size_t)D * C;              // C*C

  // Prep: weights (64K threads) and x (B*N*C/4 threads)
  prep_weights<<<dim3((C * C + 255) / 256), dim3(256), 0, stream>>>(Wq, Wk, Wv, Wqt, Wkt, Wvt);
  prep_x<<<dim3((B * N * C / 4) / 256), dim3(256), 0, stream>>>(x, xh);

  // K1: B*(N/16)*20 = 20480 waves, 2 waves/block
  qkv_kernel<<<dim3(B * (N / 16) * 20 / 2), dim3(64), 0, stream>>>(
      xh, Wqt, bq, Wkt, bk, Wvt, bv, Qb, Kb, Vt);

  // K2: B*(N/16) = 1024 waves
  row_stats_kernel<<<dim3(B * (N / 16) / 2), dim3(64), 0, stream>>>(Qb, Kb, Mr, IS);

  // K3: B*(N/16)*2 = 2048 waves
  attn_out_kernel<<<dim3(B * (N / 16) * 2 / 2), dim3(64), 0, stream>>>(
      Qb, Kb, Vt, Mr, IS, x, gm, out);
}
